// InterpretableFTTRandom_1864015807017
// MI455X (gfx1250) — matrix operand order repaired and checked by reading, not compiled, not run on hardware
//
#include <hip/hip_runtime.h>
#include <hip/hip_bf16.h>
#include <stdint.h>

typedef __attribute__((ext_vector_type(8)))  _Float16 v8h;
typedef __attribute__((ext_vector_type(16))) _Float16 v16h;
typedef __attribute__((ext_vector_type(8)))  float    v8f;

namespace ftt {
constexpr int B    = 2048;
constexpr int NN   = 24;
constexpr int NC   = 8;
constexpr int CARD = 16;
constexpr int D    = 512;
constexpr int H    = 8;
constexpr int L    = 3;
constexpr int FH   = 1024;
constexpr int S    = NN + NC + 1;   // 33
constexpr int DH   = D / H;         // 64
constexpr int M    = B * S;         // 67584 (divisible by 128)
}

__device__ __forceinline__ float wave_sum(float v) {
  #pragma unroll
  for (int o = 16; o; o >>= 1) v += __shfl_xor(v, o, 32);
  return v;
}
__device__ __forceinline__ float wave_max(float v) {
  #pragma unroll
  for (int o = 16; o; o >>= 1) v = fmaxf(v, __shfl_xor(v, o, 32));
  return v;
}

// async global->LDS staging (CDNA5 TDM-adjacent path, ASYNCcnt-tracked)
__device__ __forceinline__ void async_load_b128(unsigned lds_off, const void* gaddr) {
  asm volatile("global_load_async_to_lds_b128 %0, %1, off"
               :: "v"(lds_off), "v"((uint64_t)(uintptr_t)gaddr) : "memory");
}
__device__ __forceinline__ void wait_async0() {
  asm volatile("s_wait_asynccnt 0" ::: "memory");
}
__device__ __forceinline__ unsigned lds_offset_of(const void* p) {
  // ISA 10.2: for LDS-aperture flat addresses, the LDS offset is the low 32 bits of the address
  return (unsigned)(uintptr_t)p;
}

// ---------------------------------------------------------------------------
// Tokenizer: build x[M, D] fp32
// ---------------------------------------------------------------------------
__global__ __launch_bounds__(256)
void tokenizer_kernel(const float* __restrict__ x_num, const int* __restrict__ x_cat,
                      const float* __restrict__ w_num, const float* __restrict__ b_num,
                      const float* __restrict__ emb_cat, const float* __restrict__ b_cat,
                      const float* __restrict__ cls, float* __restrict__ x) {
  using namespace ftt;
  int idx = blockIdx.x * 256 + threadIdx.x;           // < M*D
  int m = idx >> 9;           // token row
  int d = idx & 511;
  int b = m / S;
  int s = m - b * S;
  float val;
  if (s < NN) {
    val = x_num[b * NN + s] * w_num[s * D + d] + b_num[s * D + d];
  } else if (s < NN + NC) {
    int j = s - NN;
    int c = x_cat[b * NC + j];
    val = emb_cat[(size_t)(j * CARD + c) * D + d] + b_cat[j * D + d];
  } else {
    val = cls[d];
  }
  x[idx] = val;
}

// ---------------------------------------------------------------------------
// Weight convert + transpose: in fp32 [K][N] -> out fp16 [N][K]
// ---------------------------------------------------------------------------
__global__ __launch_bounds__(256)
void convert_wT_kernel(const float* __restrict__ in, _Float16* __restrict__ out,
                       int K, int N) {
  int idx = blockIdx.x * 256 + threadIdx.x;           // < N*K
  int n = idx / K;
  int k = idx - n * K;
  out[idx] = (_Float16)in[(size_t)k * N + n];
}

// ---------------------------------------------------------------------------
// LayerNorm rows of x[rows, 512] -> fp16 out (one wave per row)
// ---------------------------------------------------------------------------
__global__ __launch_bounds__(256)
void layernorm_f16_kernel(const float* __restrict__ x, const float* __restrict__ w,
                          const float* __restrict__ b, _Float16* __restrict__ out,
                          int rows) {
  int wave = threadIdx.x >> 5, lane = threadIdx.x & 31;
  int row = blockIdx.x * 8 + wave;
  if (row >= rows) return;
  const float* xr = x + (size_t)row * 512;
  float xs[16];
  float s = 0.f;
  #pragma unroll
  for (int t = 0; t < 16; ++t) { xs[t] = xr[lane + t * 32]; s += xs[t]; }
  float mean = wave_sum(s) * (1.0f / 512.0f);
  float v = 0.f;
  #pragma unroll
  for (int t = 0; t < 16; ++t) { float d = xs[t] - mean; v += d * d; }
  float rstd = rsqrtf(wave_sum(v) * (1.0f / 512.0f) + 1e-5f);
  _Float16* orow = out + (size_t)row * 512;
  #pragma unroll
  for (int t = 0; t < 16; ++t) {
    int d = lane + t * 32;
    orow[d] = (_Float16)((xs[t] - mean) * rstd * w[d] + b[d]);
  }
}

// ---------------------------------------------------------------------------
// WMMA GEMM: C[M,N] = A16[M,K] @ Bt16[N,K]^T, fused epilogues.
// MODE 0: out fp16 = acc + bias
// MODE 1: out fp32 = resid + acc + bias   (residual add, N == 512)
// MODE 2: ReGLU: a = acc(Bt rows n) + bias[n]; g = acc(Bt rows n+FH) + bias[FH+n];
//         out fp16 [M,FH] = a * relu(g)
// Block: 256 threads (8 waves), 128x64 tile, K-step 32.
// Wave w owns a 32x32 output tile: mw = (w&3)*32, nw = (w>>2)*32,
// i.e. 2 A-frags x 2 B-frags -> 4 WMMAs per K-step (8 in ReGLU mode).
// Tiles staged with global_load_async_to_lds_b128 (ASYNCcnt).
// ---------------------------------------------------------------------------
template <int MODE>
__global__ __launch_bounds__(256)
void gemm_wmma_kernel(const _Float16* __restrict__ A, const _Float16* __restrict__ Bt,
                      const float* __restrict__ bias, const float* __restrict__ resid,
                      float* __restrict__ outF, _Float16* __restrict__ outH,
                      int N, int K) {
  using namespace ftt;
  __shared__ _Float16 sA[128 * 32];
  __shared__ _Float16 sB[64 * 32];
  __shared__ _Float16 sG[(MODE == 2) ? 64 * 32 : 8];

  const int tid  = threadIdx.x;
  const int lane = tid & 31;
  const int wave = tid >> 5;
  const int mw = (wave & 3) * 32;
  const int nw = (wave >> 2) * 32;
  const int m0 = blockIdx.y * 128;
  const int n0 = blockIdx.x * 64;

  // staging coords: thread t moves 16-byte chunks; chunk c: row c>>2, col (c&3)*8
  const int sr = tid >> 2;            // 0..63
  const int sc = (tid & 3) * 8;       // 0,8,16,24

  const unsigned ldsA0 = lds_offset_of(&sA[sr * 32 + sc]);
  const unsigned ldsA1 = lds_offset_of(&sA[(sr + 64) * 32 + sc]);
  const unsigned ldsB  = lds_offset_of(&sB[sr * 32 + sc]);
  const unsigned ldsG  = lds_offset_of(&sG[(MODE == 2) ? (sr * 32 + sc) : 0]);

  v8f c00 = 0.f, c01 = 0.f, c10 = 0.f, c11 = 0.f;
  v8f g00 = 0.f, g01 = 0.f, g10 = 0.f, g11 = 0.f;

  union HV { v16h v; v8h h[2]; };

  for (int kk = 0; kk < K; kk += 32) {
    // ---- async-stage tiles into LDS ----
    async_load_b128(ldsA0, A + (size_t)(m0 + sr) * K + kk + sc);
    async_load_b128(ldsA1, A + (size_t)(m0 + 64 + sr) * K + kk + sc);
    async_load_b128(ldsB,  Bt + (size_t)(n0 + sr) * K + kk + sc);
    if (MODE == 2)
      async_load_b128(ldsG, Bt + (size_t)(n0 + sr + FH) * K + kk + sc);
    if (kk + 32 < K) {   // prefetch next K tile -> global_prefetch_b8
      __builtin_prefetch(A  + (size_t)(m0 + sr) * K + kk + 32 + sc, 0, 1);
      __builtin_prefetch(Bt + (size_t)(n0 + sr) * K + kk + 32 + sc, 0, 1);
    }
    wait_async0();
    __syncthreads();

    // ---- load fragments (k = 16*half + s in both A and B; contiguous 16B LDS reads) ----
    const int half = lane >> 4;
    const int r16  = lane & 15;
    HV a0, a1, b0, b1, q0, q1;
    {
      const int ab0 = (mw + r16) * 32 + half * 16;  // K contiguous 16 per lane-half
      a0.h[0] = *(const v8h*)&sA[ab0];
      a0.h[1] = *(const v8h*)&sA[ab0 + 8];
      const int ab1 = (mw + 16 + r16) * 32 + half * 16;
      a1.h[0] = *(const v8h*)&sA[ab1];
      a1.h[1] = *(const v8h*)&sA[ab1 + 8];
    }
    {
      const int bb0 = (nw + r16) * 32 + half * 16;  // K contiguous 16 per lane-half
      b0.h[0] = *(const v8h*)&sB[bb0];
      b0.h[1] = *(const v8h*)&sB[bb0 + 8];
      const int bb1 = (nw + 16 + r16) * 32 + half * 16;
      b1.h[0] = *(const v8h*)&sB[bb1];
      b1.h[1] = *(const v8h*)&sB[bb1 + 8];
      if (MODE == 2) {
        q0.h[0] = *(const v8h*)&sG[bb0];
        q0.h[1] = *(const v8h*)&sG[bb0 + 8];
        q1.h[0] = *(const v8h*)&sG[bb1];
        q1.h[1] = *(const v8h*)&sG[bb1 + 8];
      }
    }

    c00 = __builtin_amdgcn_wmma_f32_16x16x32_f16(false, a0.v, false, b0.v, (short)0, c00, false, false);
    c01 = __builtin_amdgcn_wmma_f32_16x16x32_f16(false, a0.v, false, b1.v, (short)0, c01, false, false);
    c10 = __builtin_amdgcn_wmma_f32_16x16x32_f16(false, a1.v, false, b0.v, (short)0, c10, false, false);
    c11 = __builtin_amdgcn_wmma_f32_16x16x32_f16(false, a1.v, false, b1.v, (short)0, c11, false, false);
    if (MODE == 2) {
      g00 = __builtin_amdgcn_wmma_f32_16x16x32_f16(false, a0.v, false, q0.v, (short)0, g00, false, false);
      g01 = __builtin_amdgcn_wmma_f32_16x16x32_f16(false, a0.v, false, q1.v, (short)0, g01, false, false);
      g10 = __builtin_amdgcn_wmma_f32_16x16x32_f16(false, a1.v, false, q0.v, (short)0, g10, false, false);
      g11 = __builtin_amdgcn_wmma_f32_16x16x32_f16(false, a1.v, false, q1.v, (short)0, g11, false, false);
    }
    __syncthreads();
  }

  // ---- epilogue: C/D layout: lane n = lane&15; row = (lane>>4)*8 + j ----
  const int cn = lane & 15;
  const int mb0 = m0 + mw + (lane >> 4) * 8;
  const int mb1 = mb0 + 16;
  const int col0 = n0 + nw + cn;
  const int col1 = n0 + nw + 16 + cn;
  #pragma unroll
  for (int j = 0; j < 8; ++j) {
    const size_t r0 = (size_t)(mb0 + j);
    const size_t r1 = (size_t)(mb1 + j);
    if (MODE == 0) {
      outH[r0 * N + col0] = (_Float16)(c00[j] + bias[col0]);
      outH[r0 * N + col1] = (_Float16)(c01[j] + bias[col1]);
      outH[r1 * N + col0] = (_Float16)(c10[j] + bias[col0]);
      outH[r1 * N + col1] = (_Float16)(c11[j] + bias[col1]);
    } else if (MODE == 1) {
      outF[r0 * N + col0] = resid[r0 * N + col0] + c00[j] + bias[col0];
      outF[r0 * N + col1] = resid[r0 * N + col1] + c01[j] + bias[col1];
      outF[r1 * N + col0] = resid[r1 * N + col0] + c10[j] + bias[col0];
      outF[r1 * N + col1] = resid[r1 * N + col1] + c11[j] + bias[col1];
    } else {
      float a00 = c00[j] + bias[col0], w00 = g00[j] + bias[FH + col0];
      float a01 = c01[j] + bias[col1], w01 = g01[j] + bias[FH + col1];
      float a10 = c10[j] + bias[col0], w10 = g10[j] + bias[FH + col0];
      float a11 = c11[j] + bias[col1], w11 = g11[j] + bias[FH + col1];
      outH[r0 * N + col0] = (_Float16)(a00 * fmaxf(w00, 0.f));
      outH[r0 * N + col1] = (_Float16)(a01 * fmaxf(w01, 0.f));
      outH[r1 * N + col0] = (_Float16)(a10 * fmaxf(w10, 0.f));
      outH[r1 * N + col1] = (_Float16)(a11 * fmaxf(w11, 0.f));
    }
  }
}

// ---------------------------------------------------------------------------
// Sparse masked attention for one (batch, head) per block. S=33, DH=64.
// ---------------------------------------------------------------------------
__global__ __launch_bounds__(256)
void attn_kernel(const _Float16* __restrict__ q16, const _Float16* __restrict__ k16,
                 const _Float16* __restrict__ v16, const unsigned char* __restrict__ mask,
                 _Float16* __restrict__ o16) {
  using namespace ftt;
  __shared__ _Float16 sQ[S * DH];
  __shared__ _Float16 sK[S * DH];
  __shared__ _Float16 sV[S * DH];
  __shared__ float    sP[8][S + 3];

  const int tid = threadIdx.x;
  const int lane = tid & 31;
  const int wave = tid >> 5;
  const int b  = blockIdx.x / H;
  const int hh = blockIdx.x - b * H;
  const size_t rowbase = (size_t)b * S;
  const int colbase = hh * DH;

  for (int idx = tid; idx < S * DH; idx += 256) {
    int s = idx >> 6, d = idx & 63;
    size_t g = (rowbase + s) * D + colbase + d;
    sQ[idx] = q16[g];
    sK[idx] = k16[g];
    sV[idx] = v16[g];
  }
  __syncthreads();

  const float scale = 0.125f;   // 1/sqrt(64)
  for (int i = wave; i < S; i += 8) {
    // scores: lane j handles column j; lane 0 additionally column 32
    float acc = 0.f;
    #pragma unroll 8
    for (int d = 0; d < DH; ++d)
      acc += (float)sQ[i * DH + d] * (float)sK[lane * DH + d];
    float s0 = mask[i * S + lane] ? acc * scale : -1e9f;
    float s1 = -1e30f;
    if (lane == 0) {
      float acc2 = 0.f;
      #pragma unroll 8
      for (int d = 0; d < DH; ++d)
        acc2 += (float)sQ[i * DH + d] * (float)sK[32 * DH + d];
      s1 = mask[i * S + 32] ? acc2 * scale : -1e9f;
    }
    float mx = wave_max(fmaxf(s0, s1));
    float p0 = __expf(s0 - mx);
    float p1 = (lane == 0) ? __expf(s1 - mx) : 0.f;
    float inv = 1.0f / wave_sum(p0 + p1);
    sP[wave][lane] = p0 * inv;
    if (lane == 0) sP[wave][32] = p1 * inv;

    // o_i[d]: lane covers dims {lane, lane+32}
    float o0 = 0.f, o1 = 0.f;
    #pragma unroll 8
    for (int j = 0; j < S; ++j) {
      float p = sP[wave][j];
      o0 += p * (float)sV[j * DH + lane];
      o1 += p * (float)sV[j * DH + 32 + lane];
    }
    size_t g = (rowbase + i) * D + colbase;
    o16[g + lane]      = (_Float16)o0;
    o16[g + 32 + lane] = (_Float16)o1;
  }
}

// ---------------------------------------------------------------------------
// Head: LN(x[:, -1]) -> relu -> dot with Wh (DOUT==1). One wave per batch row.
// ---------------------------------------------------------------------------
__global__ __launch_bounds__(256)
void head_kernel(const float* __restrict__ x, const float* __restrict__ w,
                 const float* __restrict__ b, const float* __restrict__ Wh,
                 const float* __restrict__ bh, float* __restrict__ out) {
  using namespace ftt;
  int wave = threadIdx.x >> 5, lane = threadIdx.x & 31;
  int row = blockIdx.x * 8 + wave;             // batch index
  const float* xr = x + ((size_t)row * S + (S - 1)) * D;
  float xs[16];
  float s = 0.f;
  #pragma unroll
  for (int t = 0; t < 16; ++t) { xs[t] = xr[lane + t * 32]; s += xs[t]; }
  float mean = wave_sum(s) * (1.0f / 512.0f);
  float v = 0.f;
  #pragma unroll
  for (int t = 0; t < 16; ++t) { float d = xs[t] - mean; v += d * d; }
  float rstd = rsqrtf(wave_sum(v) * (1.0f / 512.0f) + 1e-5f);
  float acc = 0.f;
  #pragma unroll
  for (int t = 0; t < 16; ++t) {
    int d = lane + t * 32;
    float c = (xs[t] - mean) * rstd * w[d] + b[d];
    acc += fmaxf(c, 0.f) * Wh[d];
  }
  acc = wave_sum(acc);
  if (lane == 0) out[row] = acc + bh[0];
}

// ---------------------------------------------------------------------------
extern "C" void kernel_launch(void* const* d_in, const int* in_sizes, int n_in,
                              void* d_out, int out_size, void* d_ws, size_t ws_size,
                              hipStream_t stream) {
  using namespace ftt;
  const float* x_num   = (const float*)d_in[0];
  const int*   x_cat   = (const int*)d_in[1];
  const float* w_num   = (const float*)d_in[2];
  const float* b_num   = (const float*)d_in[3];
  const float* emb_cat = (const float*)d_in[4];
  const float* b_cat   = (const float*)d_in[5];
  const float* cls     = (const float*)d_in[6];
  const float* ln1_w   = (const float*)d_in[7];
  const float* ln1_b   = (const float*)d_in[8];
  const float* Wq = (const float*)d_in[9];  const float* bq = (const float*)d_in[10];
  const float* Wk = (const float*)d_in[11]; const float* bk = (const float*)d_in[12];
  const float* Wv = (const float*)d_in[13]; const float* bv = (const float*)d_in[14];
  const float* Wo = (const float*)d_in[15]; const float* bo = (const float*)d_in[16];
  const float* ln2_w = (const float*)d_in[17]; const float* ln2_b = (const float*)d_in[18];
  const float* Wf1 = (const float*)d_in[19]; const float* bf1 = (const float*)d_in[20];
  const float* Wf2 = (const float*)d_in[21]; const float* bf2 = (const float*)d_in[22];
  const unsigned char* mask = (const unsigned char*)d_in[23];
  const float* hln_w = (const float*)d_in[25]; const float* hln_b = (const float*)d_in[26];
  const float* Wh = (const float*)d_in[27];  const float* bh = (const float*)d_in[28];
  float* out = (float*)d_out;

  // ---- workspace carve-up (all chunks 256B-aligned sizes) ----
  char* ws = (char*)d_ws;
  size_t off = 0;
  auto carve = [&](size_t bytes) {
    void* p = ws + off;
    off += (bytes + 255) & ~(size_t)255;
    return p;
  };
  float*    x   = (float*)   carve((size_t)M * D * 4);
  _Float16* h16 = (_Float16*)carve((size_t)M * D * 2);
  _Float16* q16 = (_Float16*)carve((size_t)M * D * 2);
  _Float16* k16 = (_Float16*)carve((size_t)M * D * 2);
  _Float16* v16 = (_Float16*)carve((size_t)M * D * 2);
  const size_t perL = 4 * (size_t)D * D + 2 * (size_t)FH * D + (size_t)FH * D; // f16 elems
  _Float16* wT  = (_Float16*)carve((size_t)L * perL * 2);
  _Float16* o16 = h16;   // alias: h consumed by QKV GEMMs before attention writes o
  _Float16* u16 = q16;   // alias: ReGLU output spans q16+k16 (exactly M*FH*2 bytes)

  // ---- tokenizer ----
  tokenizer_kernel<<<(M * D) / 256, 256, 0, stream>>>(x_num, x_cat, w_num, b_num,
                                                      emb_cat, b_cat, cls, x);

  const dim3 gD(D / 64, M / 128);     // N=512 GEMMs
  const dim3 gF(FH / 64, M / 128);    // ReGLU GEMM (N=FH)

  for (int l = 0; l < L; ++l) {
    _Float16* qT  = wT + (size_t)l * perL;
    _Float16* kT  = qT + (size_t)D * D;
    _Float16* vT  = kT + (size_t)D * D;
    _Float16* oT  = vT + (size_t)D * D;
    _Float16* f1T = oT + (size_t)D * D;         // [2FH][D]
    _Float16* f2T = f1T + 2 * (size_t)FH * D;   // [D][FH]

    // convert+transpose weights -> fp16 [N][K]
    convert_wT_kernel<<<(D * D) / 256, 256, 0, stream>>>(Wq + (size_t)l * D * D, qT, D, D);
    convert_wT_kernel<<<(D * D) / 256, 256, 0, stream>>>(Wk + (size_t)l * D * D, kT, D, D);
    convert_wT_kernel<<<(D * D) / 256, 256, 0, stream>>>(Wv + (size_t)l * D * D, vT, D, D);
    convert_wT_kernel<<<(D * D) / 256, 256, 0, stream>>>(Wo + (size_t)l * D * D, oT, D, D);
    convert_wT_kernel<<<(2 * FH * D) / 256, 256, 0, stream>>>(Wf1 + (size_t)l * D * 2 * FH,
                                                              f1T, D, 2 * FH);
    convert_wT_kernel<<<(FH * D) / 256, 256, 0, stream>>>(Wf2 + (size_t)l * FH * D,
                                                          f2T, FH, D);

    // LN1 -> h16
    layernorm_f16_kernel<<<M / 8, 256, 0, stream>>>(x, ln1_w + l * D, ln1_b + l * D, h16, M);

    // Q/K/V projections (fp16 out)
    gemm_wmma_kernel<0><<<gD, 256, 0, stream>>>(h16, qT, bq + l * D, nullptr, nullptr, q16, D, D);
    gemm_wmma_kernel<0><<<gD, 256, 0, stream>>>(h16, kT, bk + l * D, nullptr, nullptr, k16, D, D);
    gemm_wmma_kernel<0><<<gD, 256, 0, stream>>>(h16, vT, bv + l * D, nullptr, nullptr, v16, D, D);

    // masked sparse attention -> o16 (aliases h16)
    attn_kernel<<<B * H, 256, 0, stream>>>(q16, k16, v16, mask + (size_t)l * S * S, o16);

    // x += o @ Wo + bo
    gemm_wmma_kernel<1><<<gD, 256, 0, stream>>>(o16, oT, bo + l * D, x, x, nullptr, D, D);

    // LN2 -> h16
    layernorm_f16_kernel<<<M / 8, 256, 0, stream>>>(x, ln2_w + l * D, ln2_b + l * D, h16, M);

    // ReGLU FFN first GEMM (fused a * relu(g)) -> u16
    gemm_wmma_kernel<2><<<gF, 256, 0, stream>>>(h16, f1T, bf1 + (size_t)l * 2 * FH,
                                                nullptr, nullptr, u16, FH, D);

    // x += u @ Wf2 + bf2
    gemm_wmma_kernel<1><<<gD, 256, 0, stream>>>(u16, f2T, bf2 + l * D, x, x, nullptr, D, FH);
  }

  // head on CLS token
  head_kernel<<<B / 8, 256, 0, stream>>>(x, hln_w, hln_b, Wh, bh, out);
}
